// RecurrentFFNet_67929202753577
// MI455X (gfx1250) — matrix operand order repaired and checked by reading, not compiled, not run on hardware
//
#include <hip/hip_runtime.h>

// ---------------- problem constants ----------------
constexpr int BDIM = 512;
constexpr int TDIM = 256;
constexpr int FDIM = 128;   // input features
constexpr int HDIM = 512;   // hidden
constexpr int FFD  = 1024;  // FF hidden
constexpr int ODIM = 64;    // output
constexpr float PADV = -999.0f;

// ---------------- vector types ----------------
typedef __attribute__((ext_vector_type(16))) __bf16          v16bf;
typedef __attribute__((ext_vector_type(16))) unsigned short  v16u;
typedef __attribute__((ext_vector_type(8)))  unsigned short  v8u;
typedef __attribute__((ext_vector_type(8)))  float           v8f;
typedef __attribute__((ext_vector_type(4)))  unsigned int    v4u;
typedef __attribute__((ext_vector_type(8)))  int             v8i;
typedef __attribute__((ext_vector_type(4)))  int             v4i;

// TDM availability — this toolchain exposes the 6-arg builtin
// (uint32x4 g0, int32x8 g1, int32x4 g2, int32x4 g3, int32x8 extra, i32 cpol).
#if defined(__has_builtin)
#  if __has_builtin(__builtin_amdgcn_tensor_load_to_lds) && \
      __has_builtin(__builtin_amdgcn_s_wait_tensorcnt)
#    define USE_TDM 1
#  endif
#endif
#ifndef USE_TDM
#  define USE_TDM 0
#endif

// ---------------- packed-weight workspace layout (ushort bf16 bits) ------
// B-fragment layout for v_wmma_f32_16x16x32_bf16 (K=32 x N=16 tile):
//   lane <16 : column N=lane,    K = kc*32 + 0..15   (16 contiguous values)
//   lane>=16 : column N=lane-16, K = kc*32 + 16..31
// frag index = ntile*KCHUNKS + kc ; each frag = 32 lanes * 16 elems = 512 u16.
constexpr long WS_WIH = 0;                       // 96 nt * 4  kc * 512 = 196608
constexpr long WS_WHH = WS_WIH + 196608;         // 96 nt * 16 kc * 512 = 786432
constexpr long WS_W1  = WS_WHH + 786432;         // 64 nt * 16 kc * 512 = 524288
constexpr long WS_W2  = WS_W1  + 524288;         // 4  nt * 32 kc * 512 = 65536
constexpr long WS_TOT = WS_W2  + 65536;          // 1572864 elems = 3 MB

__device__ __forceinline__ unsigned short f2bf(float f) {
  unsigned u = __builtin_bit_cast(unsigned, f);
  u += 0x7FFFu + ((u >> 16) & 1u);               // round-to-nearest-even
  return (unsigned short)(u >> 16);
}

// ---------------- weight pack kernel ----------------
__global__ __launch_bounds__(256) void pack_weights_kernel(
    const float* __restrict__ Wih,   // [3H, F]
    const float* __restrict__ Whh,   // [3H, H]
    const float* __restrict__ W1,    // [H, FF]
    const float* __restrict__ W2,    // [FF, OUT]
    unsigned short* __restrict__ ws) {
  long p = (long)blockIdx.x * blockDim.x + threadIdx.x;
  if (p >= WS_TOT) return;

  long q; int kchunks; int which;
  if      (p < WS_WHH) { q = p - WS_WIH; kchunks = 4;  which = 0; }
  else if (p < WS_W1)  { q = p - WS_WHH; kchunks = 16; which = 1; }
  else if (p < WS_W2)  { q = p - WS_W1;  kchunks = 16; which = 2; }
  else                 { q = p - WS_W2;  kchunks = 32; which = 3; }

  int  j    = (int)(q & 15);
  int  lane = (int)((q >> 4) & 31);
  long rest = q >> 9;
  int  kc   = (int)(rest % kchunks);
  int  nt   = (int)(rest / kchunks);
  int  k    = kc * 32 + ((lane >> 4) << 4) + j;  // K index inside operand
  int  n    = nt * 16 + (lane & 15);             // N index inside operand

  float v;
  switch (which) {
    case 0:  v = Wih[(long)n * FDIM + k]; break; // operand = W_ih^T (F x 3H)
    case 1:  v = Whh[(long)n * HDIM + k]; break; // operand = W_hh^T (H x 3H)
    case 2:  v = W1 [(long)k * FFD  + n]; break; // operand = W1     (H x FF)
    default: v = W2 [(long)k * ODIM + n]; break; // operand = W2     (FF x OUT)
  }
  ws[p] = f2bf(v);
}

// ---------------- fragment loaders ----------------
// A fragment (M=16 x K=32 bf16) from an LDS row-major tile.
__device__ __forceinline__ v16bf lds_a_frag(const unsigned short* lds,
                                            int stride, int k0, int lane) {
  int sel = lane >> 4;
  int row = lane & 15;
  const unsigned short* p = lds + row * stride + k0 + sel * 16;
  v8u lo = *(const v8u*)(p);
  v8u hi = *(const v8u*)(p + 8);
  v16u w;
#pragma unroll
  for (int j = 0; j < 8; ++j) { w[j] = lo[j]; w[8 + j] = hi[j]; }
  return __builtin_bit_cast(v16bf, w);
}

// B fragment: 32 contiguous bytes per lane from the packed blob.
__device__ __forceinline__ v16bf g_b_frag(const unsigned short* pk,
                                          long frag, int lane) {
  const unsigned short* p = pk + (frag * 32 + lane) * 16;
  return __builtin_bit_cast(v16bf, *(const v16u*)p);
}

__device__ __forceinline__ v8f wmma_bf16(v16bf a, v16bf b, v8f c) {
  return __builtin_amdgcn_wmma_f32_16x16x32_bf16(
      /*neg_a=*/false, a, /*neg_b=*/false, b,
      /*c_mod=*/(short)0, c, /*reuse_a=*/false, /*reuse_b=*/false);
}

__device__ __forceinline__ float sigf(float x) {
  return 1.0f / (1.0f + __expf(-x));
}

#if USE_TDM
// Issue a TDM load of a 16-row x 128-col f32 tile (row stride T*F elems)
// from global into LDS. D# per CDNA5 ISA:
//   group0: count=1 | lds_addr[63:32] | global_addr[120:64] | type=2
//   group1: data_size=4B, tensor_dim0=128, tensor_dim1=16,
//           tile_dim0=128, tile_dim1=16, tensor_dim0_stride=T*F
__device__ __forceinline__ void tdm_load_x_tile(const float* gsrc,
                                                float* lds_dst) {
  unsigned long long ga = (unsigned long long)(size_t)gsrc;
  unsigned lds_off = (unsigned)(size_t)lds_dst;   // generic LDS addr[31:0]
  v4u g0;
  g0[0] = 1u;                                      // count=1, is_restore=0
  g0[1] = lds_off;                                 // lds_addr
  g0[2] = (unsigned)ga;                            // global_addr[31:0]
  g0[3] = (unsigned)((ga >> 32) & 0x01FFFFFFull)   // global_addr[56:32]
          | (2u << 30);                            // type=2 ("image")
  v8i g1;
  g1[0] = 0x00020000;                              // data_size=2 -> 4 bytes
  g1[1] = (FDIM & 0xFFFF) << 16;                   // tensor_dim0 lo16
  g1[2] = (FDIM >> 16) | (16 << 16);               // dim0 hi | tensor_dim1 lo
  g1[3] = (FDIM << 16);                            // dim1 hi | tile_dim0
  g1[4] = 16;                                      // tile_dim1=16, tile_dim2=0
  g1[5] = TDIM * FDIM;                             // tensor_dim0_stride lo32
  g1[6] = 0;                                       // stride0 hi | stride1 lo
  g1[7] = 0;
  v4i z4 = {0, 0, 0, 0};                           // groups 2/3 unused (2D)
  v8i z8 = {0, 0, 0, 0, 0, 0, 0, 0};               // extra group (zeroed)
  __builtin_amdgcn_tensor_load_to_lds(g0, g1, z4, z4, z8, 0);
}
#endif

// ---------------- fused GRU + FF kernel ----------------
// One workgroup (256 thr = 8 waves) owns 16 batch rows, loops over all T.
constexpr int SMEM_BYTES =
    (8192 + 2048 + 8192 + 16384) * 2 +        // bf16: h, x_t, ff_in, hid
    (4096 + 32768 + 8192 + 2048 + 1024 + 64) * 4 + // f32: x stage, gates, ...
    16 * 4;                                   // seq lens  (~256 KB)

__global__ __launch_bounds__(256, 1) void gru_ff_fused_kernel(
    const float* __restrict__ x,        // [B, T, F]
    const int*   __restrict__ seq,      // [B]
    const float* __restrict__ b_ih,     // [3H]
    const float* __restrict__ b_hh,     // [3H]
    const float* __restrict__ b1,       // [FF]
    const float* __restrict__ b2,       // [OUT]
    const unsigned short* __restrict__ ws,
    float* __restrict__ out) {          // [B, T, OUT]
  extern __shared__ char smem[];
  unsigned short* sh_h  = (unsigned short*)smem;  // [16][512] h (bf16)
  unsigned short* sh_x  = sh_h  + 8192;           // [16][128] x_t (bf16)
  unsigned short* sh_ff = sh_x  + 2048;           // [16][512] ff input (bf16)
  unsigned short* sh_hd = sh_ff + 8192;           // [16][1024] hid (bf16)
  float* sh_xs   = (float*)(sh_hd + 16384);       // [2][16*128] x stage (f32)
  float* sh_g    = sh_xs   + 4096;                // [16][2048] rsum|zsum|i_n|h_n
  float* sh_hf   = sh_g    + 32768;               // [16][512] h (f32 master)
  float* sh_bias = sh_hf   + 8192;                // [2048] gate biases
  float* sh_b1   = sh_bias + 2048;                // [1024]
  float* sh_b2   = sh_b1   + 1024;                // [64]
  int*   sh_len  = (int*)(sh_b2 + 64);            // [16]

  const unsigned short* ws_wih = ws + WS_WIH;
  const unsigned short* ws_whh = ws + WS_WHH;
  const unsigned short* ws_w1  = ws + WS_W1;
  const unsigned short* ws_w2  = ws + WS_W2;

  const int tid  = threadIdx.x;
  const int lane = tid & 31;
  // readfirstlane -> SGPR: all wave-level branches become scalar branches,
  // keeping EXEC all-ones around WMMA (ISA requirement).
  const int wave = __builtin_amdgcn_readfirstlane(tid >> 5);
  const int sel  = lane >> 4;
  const int cc   = lane & 15;
  const int b0   = blockIdx.x * 16;

#if USE_TDM
  // Prefetch x tile for t=0 (wave 0 owns the TDM queue for this WG).
  if (wave == 0) tdm_load_x_tile(x + (size_t)b0 * TDIM * FDIM, sh_xs);
#endif

  // ---- one-time init: biases, seq lens, h = 0 ----
  for (int i = tid; i < 2048; i += 256) {
    float v;
    if      (i < 1024) v = b_ih[i] + b_hh[i];   // r,z gates: sum is fine
    else if (i < 1536) v = b_ih[i];             // i_n bias
    else               v = b_hh[i - 512];       // h_n bias (b_hh[1024..1535])
    sh_bias[i] = v;
  }
  for (int i = tid; i < 1024; i += 256) sh_b1[i] = b1[i];
  if (tid < 64) sh_b2[tid] = b2[tid];
  if (tid < 16) sh_len[tid] = seq[b0 + tid];
  for (int i = tid; i < 8192; i += 256) { sh_hf[i] = 0.0f; sh_h[i] = 0; }
  __syncthreads();

  for (int t = 0; t < TDIM; ++t) {
    // ---- phase 1: x_t tile -> LDS bf16 ----
#if USE_TDM
    if (wave == 0) __builtin_amdgcn_s_wait_tensorcnt(0);
    __syncthreads();                       // publish TDM-written LDS
    {
      const float* stg = sh_xs + (t & 1) * 2048;
      for (int q = tid; q < 512; q += 256) {
        int base = q * 4;
        float4 xv = *(const float4*)(stg + base);
        sh_x[base + 0] = f2bf(xv.x);
        sh_x[base + 1] = f2bf(xv.y);
        sh_x[base + 2] = f2bf(xv.z);
        sh_x[base + 3] = f2bf(xv.w);
      }
    }
    if (wave == 0 && t + 1 < TDIM)         // async prefetch next timestep
      tdm_load_x_tile(x + ((size_t)b0 * TDIM + (t + 1)) * FDIM,
                      sh_xs + ((t + 1) & 1) * 2048);
#else
    for (int q = tid; q < 512; q += 256) {
      int base = q * 4;
      int row  = base >> 7;
      int col  = base & 127;
      float4 xv = *(const float4*)(x + ((size_t)(b0 + row) * TDIM + t) * FDIM + col);
      sh_x[base + 0] = f2bf(xv.x);
      sh_x[base + 1] = f2bf(xv.y);
      sh_x[base + 2] = f2bf(xv.z);
      sh_x[base + 3] = f2bf(xv.w);
    }
#endif
    __syncthreads();

    // ---- phase 2: gates GEMM (128 col-tile tasks over 8 waves) ----
    // task 0..63  : r,z gates  acc = x@Wih^T + h@Whh^T   (20 WMMA)
    // task 64..95 : i_n        acc = x@Wih^T             (4 WMMA)
    // task 96..127: h_n        acc = h@Whh^T             (16 WMMA)
    for (int task = wave; task < 128; task += 8) {
      v8f acc = {0.f, 0.f, 0.f, 0.f, 0.f, 0.f, 0.f, 0.f};
      const int col0 = task * 16;
      if (task < 64) {
        __builtin_prefetch(ws_whh + ((long)task * 16 * 32) * 16, 0, 1);
#pragma unroll
        for (int kc = 0; kc < 4; ++kc)
          acc = wmma_bf16(lds_a_frag(sh_x, FDIM, kc * 32, lane),
                          g_b_frag(ws_wih, (long)task * 4 + kc, lane), acc);
#pragma unroll 4
        for (int kc = 0; kc < 16; ++kc)
          acc = wmma_bf16(lds_a_frag(sh_h, HDIM, kc * 32, lane),
                          g_b_frag(ws_whh, (long)task * 16 + kc, lane), acc);
      } else if (task < 96) {
#pragma unroll
        for (int kc = 0; kc < 4; ++kc)
          acc = wmma_bf16(lds_a_frag(sh_x, FDIM, kc * 32, lane),
                          g_b_frag(ws_wih, (long)task * 4 + kc, lane), acc);
      } else {
        const int nt = task - 32;   // Whh col-tile 64..95
#pragma unroll 4
        for (int kc = 0; kc < 16; ++kc)
          acc = wmma_bf16(lds_a_frag(sh_h, HDIM, kc * 32, lane),
                          g_b_frag(ws_whh, (long)nt * 16 + kc, lane), acc);
      }
#pragma unroll
      for (int g = 0; g < 8; ++g)
        sh_g[(g + sel * 8) * 2048 + col0 + cc] = acc[g];
    }
    __syncthreads();

    // ---- phase 3: GRU elementwise (16x512) ----
    for (int it = 0; it < 32; ++it) {
      int e   = tid + it * 256;
      int row = e >> 9;
      int col = e & 511;
      const float* gr = sh_g + row * 2048;
      float r_  = sigf(gr[col]        + sh_bias[col]);
      float z_  = sigf(gr[512 + col]  + sh_bias[512 + col]);
      float in_ =      gr[1024 + col] + sh_bias[1024 + col];
      float hn_ =      gr[1536 + col] + sh_bias[1536 + col];
      float n_  = tanhf(in_ + r_ * hn_);
      float hold = sh_hf[row * 512 + col];
      float hnew = (1.0f - z_) * n_ + z_ * hold;
      bool  vld  = (t < sh_len[row]);
      float hkeep = vld ? hnew : hold;
      sh_hf[row * 512 + col] = hkeep;
      sh_h [row * 512 + col] = f2bf(hkeep);
      sh_ff[row * 512 + col] = vld ? f2bf(hnew) : (unsigned short)0;
    }
    __syncthreads();

    // ---- phase 4: FF1  hid = relu(ff_in @ W1 + b1)  (64 tiles / 8 waves) ----
    for (int nt = wave; nt < 64; nt += 8) {
      v8f acc = {0.f, 0.f, 0.f, 0.f, 0.f, 0.f, 0.f, 0.f};
#pragma unroll 4
      for (int kc = 0; kc < 16; ++kc)
        acc = wmma_bf16(lds_a_frag(sh_ff, HDIM, kc * 32, lane),
                        g_b_frag(ws_w1, (long)nt * 16 + kc, lane), acc);
#pragma unroll
      for (int g = 0; g < 8; ++g) {
        float v = acc[g] + sh_b1[nt * 16 + cc];
        sh_hd[(g + sel * 8) * 1024 + nt * 16 + cc] = f2bf(v > 0.f ? v : 0.f);
      }
    }
    __syncthreads();

    // ---- phase 5: FF2  y = hid @ W2 + b2, mask -> global (4 tiles, waves 0-3)
    if (wave < 4) {
      const int nt = wave;
      v8f acc = {0.f, 0.f, 0.f, 0.f, 0.f, 0.f, 0.f, 0.f};
#pragma unroll 4
      for (int kc = 0; kc < 32; ++kc)
        acc = wmma_bf16(lds_a_frag(sh_hd, FFD, kc * 32, lane),
                        g_b_frag(ws_w2, (long)nt * 32 + kc, lane), acc);
#pragma unroll
      for (int g = 0; g < 8; ++g) {
        int   row = g + sel * 8;
        float yv  = (t < sh_len[row]) ? (acc[g] + sh_b2[nt * 16 + cc]) : PADV;
        out[((size_t)(b0 + row) * TDIM + t) * ODIM + nt * 16 + cc] = yv;
      }
    }
    __syncthreads();
  }
}

// ---------------- launcher ----------------
extern "C" void kernel_launch(void* const* d_in, const int* in_sizes, int n_in,
                              void* d_out, int out_size, void* d_ws, size_t ws_size,
                              hipStream_t stream) {
  const float* x    = (const float*)d_in[0];
  const int*   seq  = (const int*)  d_in[1];
  const float* Wih  = (const float*)d_in[2];
  const float* Whh  = (const float*)d_in[3];
  const float* bih  = (const float*)d_in[4];
  const float* bhh  = (const float*)d_in[5];
  const float* W1   = (const float*)d_in[6];
  const float* b1   = (const float*)d_in[7];
  const float* W2   = (const float*)d_in[8];
  const float* b2   = (const float*)d_in[9];
  float*       out  = (float*)d_out;
  unsigned short* ws = (unsigned short*)d_ws;

  // Pack weights to bf16 in WMMA B-fragment order (~3 MB, L2-resident).
  pack_weights_kernel<<<(int)((WS_TOT + 255) / 256), 256, 0, stream>>>(
      Wih, Whh, W1, W2, ws);

  // Fused recurrent + FF kernel: 32 WGs x 16 batch rows, loops T internally.
  gru_ff_fused_kernel<<<BDIM / 16, 256, SMEM_BYTES, stream>>>(
      x, seq, bih, bhh, b1, b2, ws, out);
}
